// GRULayer_28424093565438
// MI455X (gfx1250) — matrix operand order repaired and checked by reading, not compiled, not run on hardware
//
#include <hip/hip_runtime.h>
#include <hip/hip_bf16.h>

// ---------------------------------------------------------------------------
// GRU layer for MI455X (gfx1250, wave32, WMMA).
//   Phase 0: transpose+convert weights to K-major bf16 (tiny).
//   Phase 1: gi = x @ [Wir|Wiz|Win] + bias  -- huge parallel bf16-WMMA GEMM,
//            result stored bf16 in workspace, layout [b*T+t][384].
//   Phase 2: sequential GRU recurrence. 4 workgroups (16 batch rows each),
//            24 waves/WG = one 16x16 WMMA N-tile per wave. Recurrent weights
//            live in registers; hidden state lives in LDS (fp32 master +
//            bf16 operand copy); the per-step gi tile is double-buffered in
//            LDS via async global->LDS DMA (ASYNCcnt), staged one step ahead.
// ---------------------------------------------------------------------------

typedef __attribute__((ext_vector_type(16))) __bf16 v16bf;
typedef __attribute__((ext_vector_type(8)))  __bf16 v8bf;
typedef __attribute__((ext_vector_type(8)))  float  v8f;
typedef __attribute__((ext_vector_type(4)))  int    v4i;

// address-space-qualified int4 pointers for the async global->LDS builtin
// (per compiler diagnostic: params are int4 pointers in AS1 / AS3)
typedef v4i __attribute__((address_space(1))) as1_v4i;
typedef v4i __attribute__((address_space(3))) as3_v4i;

#define B_   64
#define T_   4096
#define K_   128     // D_IN == HIDDEN
#define N3_  384     // 3 * HIDDEN (r | z | n)

#if defined(__has_builtin)
#if __has_builtin(__builtin_amdgcn_global_load_async_to_lds_b128)
#define USE_ASYNC_LDS 1
#endif
#endif
#ifndef USE_ASYNC_LDS
#define USE_ASYNC_LDS 0
#endif

__device__ __forceinline__ float sigmoid_(float x) {
    return 1.0f / (1.0f + __expf(-x));
}
__device__ __forceinline__ float tanh_(float x) {
    // tanh(x) = 1 - 2/(exp(2x)+1)
    return 1.0f - 2.0f / (__expf(2.0f * x) + 1.0f);
}

// --------------------------------------------------------------------------
// Phase 0: weights -> K-major bf16.  wih_bf[n][k], whh_bf[n][k], n in [0,384)
// matches the B-fragment layout: lane n, group g reads K = kc*32+g*8..+7 and kc*32+16+g*8..+7
// (two 16B loads).
// --------------------------------------------------------------------------
__global__ void gru_prep(const float* __restrict__ Wir, const float* __restrict__ Wiz,
                         const float* __restrict__ Win, const float* __restrict__ Whr,
                         const float* __restrict__ Whz, const float* __restrict__ Whn,
                         __bf16* __restrict__ wih_bf, __bf16* __restrict__ whh_bf) {
    int n = blockIdx.x;           // 0..383  output column (gate-major)
    int k = threadIdx.x;          // 0..127
    int gate = n >> 7;
    int c = n & 127;
    const float* Wi = (gate == 0) ? Wir : (gate == 1) ? Wiz : Win;
    const float* Wh = (gate == 0) ? Whr : (gate == 1) ? Whz : Whn;
    wih_bf[(size_t)n * K_ + k] = (__bf16)Wi[k * 128 + c];
    whh_bf[(size_t)n * K_ + k] = (__bf16)Wh[k * 128 + c];
}

// --------------------------------------------------------------------------
// Phase 1: gi[row][n] = x[row] @ Wih[:,n] + b_i[n], row = b*T+t (262144 rows).
// One wave per 16x16 output tile, 8 waves per block.
// --------------------------------------------------------------------------
__global__ void __launch_bounds__(256)
gru_gemm_in(const float* __restrict__ x,
            const __bf16* __restrict__ wih_bf,
            const float* __restrict__ b_ir, const float* __restrict__ b_iz,
            const float* __restrict__ b_in,
            __bf16* __restrict__ gi) {
    const int lane  = threadIdx.x & 31;
    const int wave  = threadIdx.x >> 5;
    const int tile  = blockIdx.x * 8 + wave;     // 0 .. 393215
    const int ntile = tile % 24;
    const int mtile = tile / 24;
    const int g  = lane >> 4;                    // lane group (0/1)
    const int lm = lane & 15;                    // A: row within tile; B/D: col
    const int row0 = mtile * 16;

    v8f acc = {};
#pragma unroll
    for (int kc = 0; kc < 4; ++kc) {
        // ---- A fragment: x rows (fp32 -> bf16 on the fly) -----------------
        const float* xa = x + (size_t)(row0 + lm) * K_ + kc * 32 + g * 8;
        float4 a0 = *(const float4*)(xa);        // K = base .. base+3
        float4 a1 = *(const float4*)(xa + 4);    // K = base+4 .. base+7
        float4 a2 = *(const float4*)(xa + 16);   // K = base+16 .. base+19
        float4 a3 = *(const float4*)(xa + 20);   // K = base+20 .. base+23
        v16bf a;
        a[0]  = (__bf16)a0.x; a[1]  = (__bf16)a0.y; a[2]  = (__bf16)a0.z; a[3]  = (__bf16)a0.w;
        a[4]  = (__bf16)a1.x; a[5]  = (__bf16)a1.y; a[6]  = (__bf16)a1.z; a[7]  = (__bf16)a1.w;
        a[8]  = (__bf16)a2.x; a[9]  = (__bf16)a2.y; a[10] = (__bf16)a2.z; a[11] = (__bf16)a2.w;
        a[12] = (__bf16)a3.x; a[13] = (__bf16)a3.y; a[14] = (__bf16)a3.z; a[15] = (__bf16)a3.w;
        // ---- B fragment: K-major bf16 weights -----------------------------
        const __bf16* bp = wih_bf + (size_t)(ntile * 16 + lm) * K_ + kc * 32 + g * 8;
        v8bf blo = *(const v8bf*)(bp);
        v8bf bhi = *(const v8bf*)(bp + 16);
        v16bf b;
#pragma unroll
        for (int i = 0; i < 8; ++i) { b[i] = blo[i]; b[i + 8] = bhi[i]; }
        acc = __builtin_amdgcn_wmma_f32_16x16x32_bf16(false, a, false, b,
                                                      (short)0, acc, false, false);
    }

    // bias for this column (uniform over the 8 accumulator rows)
    const int n0 = ntile * 16 + lm;
    const int gate = n0 >> 7, c = n0 & 127;
    const float bias = (gate == 0 ? b_ir : gate == 1 ? b_iz : b_in)[c];

#pragma unroll
    for (int j = 0; j < 8; ++j) {
        int row = row0 + g * 8 + j;              // C/D layout: VGPR j -> row g*8+j
        gi[(size_t)row * N3_ + n0] = (__bf16)(acc[j] + bias);
    }
}

// --------------------------------------------------------------------------
// Phase 2: sequential recurrence. blockIdx.x in [0,4): 16 batch rows each.
// --------------------------------------------------------------------------
__global__ void __launch_bounds__(768)
gru_seq(const __bf16* __restrict__ gi,
        const __bf16* __restrict__ whh_bf,
        const float* __restrict__ b_hn,
        float* __restrict__ out) {
    __shared__ __align__(16) float  h_lds[16 * 128];   // fp32 master state
    __shared__ __align__(16) __bf16 hbf  [16 * 128];   // bf16 WMMA operand copy
    __shared__ __align__(16) float  scr  [16 * 384];   // h @ Whh result
    __shared__ __align__(16) __bf16 gbuf [2][16 * 384];// gi tile double buffer
    __shared__ float bhn_l[128];

    const int tid  = threadIdx.x;
    const int lane = tid & 31;
    const int wave = tid >> 5;          // 0..23 -> N-tile
    const int g    = lane >> 4;
    const int lm   = lane & 15;
    const int b0   = blockIdx.x * 16;   // first batch row of this workgroup

    // Each lane owns one 16B chunk of the 12288B gi tile (24 waves * 512B).
    const int coff = wave * 512 + lane * 16;     // byte offset within tile
    const int crow = coff / 768;                 // 768B per (row, 384 bf16)
    const int crem = coff - crow * 768;          // never straddles a row

#if USE_ASYNC_LDS
    // prologue: stage tile t=0 into gbuf[0]
    {
        const char* gsrc = (const char*)gi + ((size_t)(b0 + crow) * T_ + 0) * 768 + crem;
        char* ldst = (char*)&gbuf[0][0] + coff;
        __builtin_amdgcn_global_load_async_to_lds_b128(
            (as1_v4i*)gsrc, (as3_v4i*)ldst, 0, 0);
    }
#endif

    for (int i = tid; i < 16 * 128; i += 768) { h_lds[i] = 0.0f; hbf[i] = (__bf16)0.0f; }
    if (tid < 128) bhn_l[tid] = b_hn[tid];
    __syncthreads();

    // Preload this wave's weight fragments (constant across all 4096 steps).
    v16bf Bf[4];
    const int ncol = wave * 16 + lm;
#pragma unroll
    for (int kc = 0; kc < 4; ++kc) {
        const __bf16* bp = whh_bf + (size_t)ncol * K_ + kc * 32 + g * 8;
        v8bf blo = *(const v8bf*)(bp);
        v8bf bhi = *(const v8bf*)(bp + 16);
#pragma unroll
        for (int i = 0; i < 8; ++i) { Bf[kc][i] = blo[i]; Bf[kc][i + 8] = bhi[i]; }
    }

    for (int t = 0; t < T_; ++t) {
        const int buf = t & 1;
        // ---- h[16x128] @ Whh[128x384] : 4 bf16 WMMAs per wave -------------
        v8f acc = {};
#pragma unroll
        for (int kc = 0; kc < 4; ++kc) {
            const __bf16* ap = &hbf[lm * 128 + kc * 32 + g * 8];
            v8bf alo = *(const v8bf*)(ap);        // K = base .. base+7
            v8bf ahi = *(const v8bf*)(ap + 16);   // K = base+16 .. base+23
            v16bf a;
#pragma unroll
            for (int i = 0; i < 8; ++i) { a[i] = alo[i]; a[i + 8] = ahi[i]; }
            acc = __builtin_amdgcn_wmma_f32_16x16x32_bf16(false, a, false, Bf[kc],
                                                          (short)0, acc, false, false);
        }
#pragma unroll
        for (int j = 0; j < 8; ++j)
            scr[(g * 8 + j) * 384 + wave * 16 + lm] = acc[j];

#if USE_ASYNC_LDS
        // ---- stage gi tile for step t+1 into the other buffer -------------
        if (t + 1 < T_) {
            const char* gsrc = (const char*)gi +
                ((size_t)(b0 + crow) * T_ + (t + 1)) * 768 + crem;
            char* ldst = (char*)&gbuf[buf ^ 1][0] + coff;
            __builtin_amdgcn_global_load_async_to_lds_b128(
                (as1_v4i*)gsrc, (as3_v4i*)ldst, 0, 0);
            // warm L2 for the DMA two steps out (global_prefetch_b8)
            if (t + 2 < T_)
                __builtin_prefetch((const char*)gi +
                    ((size_t)(b0 + crow) * T_ + (t + 2)) * 768 + crem, 0, 1);
            // tile t is the older of the (at most) two outstanding DMAs
            asm volatile("s_wait_asynccnt 0x1" ::: "memory");
        } else {
            asm volatile("s_wait_asynccnt 0x0" ::: "memory");
        }
#endif
        __syncthreads();

        // ---- elementwise GRU cell ----------------------------------------
        for (int i = tid; i < 16 * 128; i += 768) {
            int m = i >> 7;
            int c = i & 127;
#if USE_ASYNC_LDS
            const __bf16* gp = &gbuf[buf][m * 384];
#else
            const __bf16* gp = gi + ((size_t)(b0 + m) * T_ + t) * N3_;
#endif
            float gr = (float)gp[c]       + scr[m * 384 + c];
            float gz = (float)gp[128 + c] + scr[m * 384 + 128 + c];
            float gn = (float)gp[256 + c];
            float hw = scr[m * 384 + 256 + c] + bhn_l[c];
            float hp = h_lds[i];
            float r  = sigmoid_(gr);
            float z  = sigmoid_(gz);
            float nn = tanh_(gn + r * hw);
            float hn = (1.0f - z) * nn + z * hp;
            h_lds[i] = hn;
            hbf[i]   = (__bf16)hn;
            out[((size_t)(b0 + m) * T_ + t) * 128 + c] = hn;
#if !USE_ASYNC_LDS
            __builtin_prefetch(gp + N3_ + c, 0, 0);
#endif
        }
        __syncthreads();
    }
}

// --------------------------------------------------------------------------
extern "C" void kernel_launch(void* const* d_in, const int* in_sizes, int n_in,
                              void* d_out, int out_size, void* d_ws, size_t ws_size,
                              hipStream_t stream) {
    const float* x    = (const float*)d_in[0];
    const float* Wir  = (const float*)d_in[1];
    const float* Wiz  = (const float*)d_in[2];
    const float* Win  = (const float*)d_in[3];
    const float* Whr  = (const float*)d_in[4];
    const float* Whz  = (const float*)d_in[5];
    const float* Whn  = (const float*)d_in[6];
    const float* b_ir = (const float*)d_in[7];
    const float* b_iz = (const float*)d_in[8];
    const float* b_in = (const float*)d_in[9];
    const float* b_hn = (const float*)d_in[10];
    float* out = (float*)d_out;

    // workspace layout: gi (bf16, 64*4096*384 = 201.3 MB) | whh_bf | wih_bf
    char*   ws       = (char*)d_ws;
    size_t  gi_bytes = (size_t)B_ * T_ * N3_ * sizeof(__bf16);
    __bf16* gi       = (__bf16*)ws;
    __bf16* whh_bf   = (__bf16*)(ws + gi_bytes);
    __bf16* wih_bf   = (__bf16*)(ws + gi_bytes + (size_t)N3_ * K_ * sizeof(__bf16));

    gru_prep<<<N3_, K_, 0, stream>>>(Wir, Wiz, Win, Whr, Whz, Whn, wih_bf, whh_bf);

    // (B*T/16) M-tiles * 24 N-tiles, 8 wave-tiles per 256-thread block
    int n_tiles = (B_ * T_ / 16) * 24;
    gru_gemm_in<<<n_tiles / 8, 256, 0, stream>>>(x, wih_bf, b_ir, b_iz, b_in, gi);

    gru_seq<<<B_ / 16, 768, 0, stream>>>(gi, whh_bf, b_hn, out);
}
